// SelfAttentionMasked_62749472194710
// MI455X (gfx1250) — matrix operand order repaired and checked by reading, not compiled, not run on hardware
//
#include <hip/hip_runtime.h>

typedef __attribute__((ext_vector_type(16))) _Float16 v16h;
typedef __attribute__((ext_vector_type(8)))  _Float16 v8h;
typedef __attribute__((ext_vector_type(8)))  float    v8f;

#define C_DIM   384
#define H_DIM   64
#define T_DIM   256
#define B_DIM   512
#define WPITCH  392   // padded C pitch for transposed weights in LDS (f16)
#define KPITCH  72    // padded H pitch for K tile in LDS (f16)
#define VPITCH  264   // padded T pitch for V^T tile in LDS (f16)
#define PPITCH  264   // padded T pitch for P tile in LDS (f16)

__device__ inline v16h pack16(v8h lo, v8h hi) {
    v16h r;
#pragma unroll
    for (int j = 0; j < 8; ++j) { r[j] = lo[j]; r[8 + j] = hi[j]; }
    return r;
}

// Async copy of 16 bytes/lane from global memory directly into LDS
// (GLOBAL_LOAD_ASYNC_TO_LDS_B128, tracked by ASYNCcnt).
__device__ inline void async_copy_b128(void* lds_dst, const void* gsrc) {
    unsigned lds_addr = (unsigned)(unsigned long long)lds_dst;  // LDS aperture: low 32 bits
    asm volatile("global_load_async_to_lds_b128 %0, %1, off"
                 :: "v"(lds_addr), "v"(gsrc)
                 : "memory");
}

__device__ inline void wait_asynccnt0() {
    asm volatile("s_wait_asynccnt 0x0" ::: "memory");
}

// ---------------------------------------------------------------------------
// Kernel 1: Q/K/V projection.  x[B*T][C] fp32 -> Qh,Kh row-major f16 [B*T][H],
// V stored transposed f16 Vt[B][H][T].  Q pre-scaled by C^-0.5.
// One wave computes a 16-row x 192-col (3 outputs x 64) tile.
// ---------------------------------------------------------------------------
__global__ __launch_bounds__(256)
void proj_qkv_kernel(const float* __restrict__ x,
                     const float* __restrict__ Wq,
                     const float* __restrict__ Wk,
                     const float* __restrict__ Wv,
                     _Float16* __restrict__ Qh,
                     _Float16* __restrict__ Kh,
                     _Float16* __restrict__ Vt)
{
    extern __shared__ _Float16 lds[];   // [3][64][WPITCH] transposed f16 weights

    const int tid = threadIdx.x;

    // Cooperative transposed weight load: W[c][h] (coalesced read) -> lds[o][h][c]
    const float* Ws[3] = { Wq, Wk, Wv };
#pragma unroll
    for (int o = 0; o < 3; ++o) {
        const float* W = Ws[o];
        for (int i = tid; i < C_DIM * H_DIM; i += 256) {
            int c = i >> 6;        // row of W
            int h = i & 63;        // col of W
            lds[(o * H_DIM + h) * WPITCH + c] = (_Float16)W[i];
        }
    }
    __syncthreads();

    const int wave = tid >> 5;
    const int lane = tid & 31;
    const int lrow = lane & 15;
    const int half = lane >> 4;

    const long rowbase = (long)blockIdx.x * 128 + (long)wave * 16;
    const long row     = rowbase + lrow;
    const float* xrow  = x + row * C_DIM;

    v8f acc[3][4];
#pragma unroll
    for (int o = 0; o < 3; ++o)
#pragma unroll
        for (int nt = 0; nt < 4; ++nt)
#pragma unroll
            for (int r = 0; r < 8; ++r) acc[o][nt][r] = 0.0f;

    for (int kc = 0; kc < C_DIM; kc += 32) {
        // A fragment (x rows, f32 -> f16).  Layout: e<8 -> K=kc+8*half+e,
        // e>=8 -> K=kc+16+8*half+(e-8).
        v8f p0 = *(const v8f*)(xrow + kc + half * 8);
        v8f p1 = *(const v8f*)(xrow + kc + 16 + half * 8);
        v16h a;
#pragma unroll
        for (int j = 0; j < 8; ++j) {
            a[j]     = (_Float16)p0[j];
            a[8 + j] = (_Float16)p1[j];
        }
#pragma unroll
        for (int o = 0; o < 3; ++o) {
#pragma unroll
            for (int nt = 0; nt < 4; ++nt) {
                // B fragment: lane = column n, j<8 -> K = kc + 8*half + j, j>=8 -> K = kc + 16 + 8*half + (j-8)
                const _Float16* bp =
                    &lds[(o * H_DIM + nt * 16 + lrow) * WPITCH + kc + half * 8];
                v16h b = pack16(*(const v8h*)bp, *(const v8h*)(bp + 16));
                acc[o][nt] = __builtin_amdgcn_wmma_f32_16x16x32_f16(
                    false, a, false, b, (short)0, acc[o][nt], false, false);
            }
        }
    }

    const float scale = 0.05103103630798287f;   // 384^-0.5 (reference uses C^-0.5)
#pragma unroll
    for (int nt = 0; nt < 4; ++nt) {
        const int h = nt * 16 + lrow;
#pragma unroll
        for (int r = 0; r < 8; ++r) {
            const long rg = rowbase + r + half * 8;   // D layout: lanes>=16 hold M=8+r
            Qh[rg * H_DIM + h] = (_Float16)(acc[0][nt][r] * scale);
            Kh[rg * H_DIM + h] = (_Float16)(acc[1][nt][r]);
            const long bb = rg >> 8;                  // batch
            const long tt = rg & 255;                 // position
            Vt[(bb * H_DIM + h) * T_DIM + tt] = (_Float16)(acc[2][nt][r]);
        }
    }
}

// ---------------------------------------------------------------------------
// Kernel 2: fused causal attention per batch.  One 256-thread WG per batch;
// K and V^T staged in LDS via GLOBAL_LOAD_ASYNC_TO_LDS_B128; each wave
// handles two 16-row query tiles.
// ---------------------------------------------------------------------------
__global__ __launch_bounds__(256)
void attn_kernel(const _Float16* __restrict__ Qh,
                 const _Float16* __restrict__ Kh,
                 const _Float16* __restrict__ Vt,
                 float* __restrict__ out)
{
    extern __shared__ _Float16 sm[];
    _Float16* Ks = sm;                          // [256][KPITCH]
    _Float16* Vs = Ks + T_DIM * KPITCH;         // [64][VPITCH]
    _Float16* Ps = Vs + H_DIM * VPITCH;         // [8 waves][16][PPITCH]

    const int b   = blockIdx.x;
    const int tid = threadIdx.x;

    // Stage K (row-major) into LDS — async, no VGPR round-trip
    const _Float16* Kg = Kh + (long)b * T_DIM * H_DIM;
    for (int i = tid; i < T_DIM * H_DIM / 8; i += 256) {
        int r = i >> 3, c8 = i & 7;
        async_copy_b128(&Ks[r * KPITCH + c8 * 8], &Kg[r * H_DIM + c8 * 8]);
    }
    // Stage V^T into LDS — async
    const _Float16* Vg = Vt + (long)b * H_DIM * T_DIM;
    for (int i = tid; i < H_DIM * T_DIM / 8; i += 256) {
        int r = i >> 5, c8 = i & 31;
        async_copy_b128(&Vs[r * VPITCH + c8 * 8], &Vg[r * T_DIM + c8 * 8]);
    }
    wait_asynccnt0();        // own wave's async transfers have landed in LDS
    __syncthreads();         // everyone's have

    const int wave = tid >> 5;
    const int lane = tid & 31;
    const int lrow = lane & 15;
    const int half = lane >> 4;
    const float NEG = -3.0e38f;

    _Float16* Pw = Ps + wave * 16 * PPITCH;     // this wave's P scratch

    for (int qi = 0; qi < 2; ++qi) {
        const int qt = wave + qi * 8;           // query tile 0..15

        // Q A-fragments (already scaled by C^-0.5)
        const _Float16* qrow = Qh + ((long)b * T_DIM + qt * 16 + lrow) * H_DIM;
        v16h aq0 = pack16(*(const v8h*)(qrow + half * 8),
                          *(const v8h*)(qrow + 16 + half * 8));
        v16h aq1 = pack16(*(const v8h*)(qrow + 32 + half * 8),
                          *(const v8h*)(qrow + 48 + half * 8));

        // ---- scores S = Q K^T with causal mask ----
        v8f s[16];
#pragma unroll
        for (int kt = 0; kt < 16; ++kt) {
            if (kt <= qt) {                     // wave-uniform: EXEC stays full
                const _Float16* kp = &Ks[(kt * 16 + lrow) * KPITCH + half * 8];
                v16h b0 = pack16(*(const v8h*)kp,        *(const v8h*)(kp + 16));
                v16h b1 = pack16(*(const v8h*)(kp + 32), *(const v8h*)(kp + 48));
                v8f c;
#pragma unroll
                for (int r = 0; r < 8; ++r) c[r] = 0.0f;
                c = __builtin_amdgcn_wmma_f32_16x16x32_f16(
                        false, aq0, false, b0, (short)0, c, false, false);
                c = __builtin_amdgcn_wmma_f32_16x16x32_f16(
                        false, aq1, false, b1, (short)0, c, false, false);
                if (kt == qt) {                 // diagonal tile: row >= col keeps
#pragma unroll
                    for (int r = 0; r < 8; ++r) {
                        const int rl = r + half * 8;
                        c[r] = (rl < lrow) ? NEG : c[r];
                    }
                }
                s[kt] = c;
            } else {
                v8f c;
#pragma unroll
                for (int r = 0; r < 8; ++r) c[r] = NEG;
                s[kt] = c;
            }
        }

        // ---- softmax over 256 keys (per row r within each 16-lane half) ----
        float inv[8];
#pragma unroll
        for (int r = 0; r < 8; ++r) {
            float m = s[0][r];
#pragma unroll
            for (int kt = 1; kt < 16; ++kt) m = fmaxf(m, s[kt][r]);
            m = fmaxf(m, __shfl_xor(m, 1, 32));
            m = fmaxf(m, __shfl_xor(m, 2, 32));
            m = fmaxf(m, __shfl_xor(m, 4, 32));
            m = fmaxf(m, __shfl_xor(m, 8, 32));
            float l = 0.0f;
#pragma unroll
            for (int kt = 0; kt < 16; ++kt) {
                float e = __expf(s[kt][r] - m);   // exp(NEG - m) == 0
                s[kt][r] = e;
                l += e;
            }
            l += __shfl_xor(l, 1, 32);
            l += __shfl_xor(l, 2, 32);
            l += __shfl_xor(l, 4, 32);
            l += __shfl_xor(l, 8, 32);
            inv[r] = 1.0f / l;
        }

        // ---- write P to LDS (row-major f16), wave-local so no barrier ----
#pragma unroll
        for (int kt = 0; kt < 16; ++kt)
#pragma unroll
            for (int r = 0; r < 8; ++r)
                Pw[(r + half * 8) * PPITCH + kt * 16 + lrow] = (_Float16)s[kt][r];

        // ---- O = P V ----
        v8f oacc[4];
#pragma unroll
        for (int nt = 0; nt < 4; ++nt)
#pragma unroll
            for (int r = 0; r < 8; ++r) oacc[nt][r] = 0.0f;

        const int kcmax = qt >> 1;              // last 32-key chunk with nonzero P
#pragma unroll
        for (int kc = 0; kc < 8; ++kc) {
            if (kc <= kcmax) {                  // wave-uniform
                const _Float16* pp = &Pw[lrow * PPITCH + kc * 32 + half * 16];
                v16h ap = pack16(*(const v8h*)pp, *(const v8h*)(pp + 8));
#pragma unroll
                for (int nt = 0; nt < 4; ++nt) {
                    const _Float16* vp =
                        &Vs[(nt * 16 + lrow) * VPITCH + kc * 32 + half * 16];
                    v16h bv = pack16(*(const v8h*)vp, *(const v8h*)(vp + 8));
                    oacc[nt] = __builtin_amdgcn_wmma_f32_16x16x32_f16(
                        false, ap, false, bv, (short)0, oacc[nt], false, false);
                }
            }
        }

        // ---- scale by 1/rowsum and store fp32 output ----
        float* orow = out + ((long)b * T_DIM + qt * 16) * H_DIM;
#pragma unroll
        for (int nt = 0; nt < 4; ++nt)
#pragma unroll
            for (int r = 0; r < 8; ++r)
                orow[(r + half * 8) * H_DIM + nt * 16 + lrow] = oacc[nt][r] * inv[r];
    }
}

// ---------------------------------------------------------------------------
extern "C" void kernel_launch(void* const* d_in, const int* in_sizes, int n_in,
                              void* d_out, int out_size, void* d_ws, size_t ws_size,
                              hipStream_t stream) {
    (void)in_sizes; (void)n_in; (void)out_size; (void)ws_size;
    const float* x  = (const float*)d_in[0];
    const float* Wq = (const float*)d_in[1];
    const float* Wk = (const float*)d_in[2];
    const float* Wv = (const float*)d_in[3];
    float* out = (float*)d_out;

    const size_t nrows = (size_t)B_DIM * T_DIM;            // 131072
    _Float16* Qh = (_Float16*)d_ws;
    _Float16* Kh = Qh + nrows * H_DIM;
    _Float16* Vt = Kh + nrows * H_DIM;                     // total 48 MiB f16

    const size_t lds1 = (size_t)3 * H_DIM * WPITCH * sizeof(_Float16);   // ~147 KB
    proj_qkv_kernel<<<dim3(nrows / 128), dim3(256), lds1, stream>>>(
        x, Wq, Wk, Wv, Qh, Kh, Vt);

    const size_t lds2 = (size_t)(T_DIM * KPITCH + H_DIM * VPITCH +
                                 8 * 16 * PPITCH) * sizeof(_Float16);    // ~135 KB
    attn_kernel<<<dim3(B_DIM), dim3(256), lds2, stream>>>(Qh, Kh, Vt, out);
}
